// SelfAttention_1228360647092
// MI455X (gfx1250) — matrix operand order repaired and checked by reading, not compiled, not run on hardware
//
#include <hip/hip_runtime.h>

// Self-attention for MI455X (gfx1250): bf16 WMMA (16x16x32, f32 accum),
// flash-attention (no materialized 256MB score tensor), wave32 layouts
// per CDNA5 ISA 7.12.2. x/W pre-converted to bf16 once so the GEMM hot
// loops are pure load+WMMA (no per-tile cvt VALU work, no WAR nop padding).

#define BATCH  4
#define SEQ    4096
#define DMODEL 512
#define EOUT   1536   // 3*DMODEL

typedef __attribute__((ext_vector_type(16))) __bf16 v16bf;
typedef __attribute__((ext_vector_type(8)))  __bf16 v8bf;
typedef __attribute__((ext_vector_type(8)))  float  v8f;

// ---------------------------------------------------------------------------
// Kernel 0: elementwise f32 -> bf16 (RNE via v_cvt_pk_bf16_f32).
// 8 elements per thread: 2x float4 load -> 16B bf16 store.
// ---------------------------------------------------------------------------
__global__ __launch_bounds__(256) void cvt_bf16_kernel(
    const float* __restrict__ src, __bf16* __restrict__ dst, int n8) {
  int i = blockIdx.x * blockDim.x + threadIdx.x;
  if (i >= n8) return;
  const float* s = src + (size_t)i * 8;
  float4 a = *(const float4*)(s);
  float4 b = *(const float4*)(s + 4);
  v8bf r;
  r[0]=(__bf16)a.x; r[1]=(__bf16)a.y; r[2]=(__bf16)a.z; r[3]=(__bf16)a.w;
  r[4]=(__bf16)b.x; r[5]=(__bf16)b.y; r[6]=(__bf16)b.z; r[7]=(__bf16)b.w;
  *(v8bf*)(dst + (size_t)i * 8) = r;
}

// One WMMA step of the QKV GEMM, bf16 operands straight from memory.
// A-frag (16x32 bf16): lane<16 -> halves 0..7: K=kk+j, 8..15: K=kk+8+j
//                      lane>=16 -> halves 0..7: K=kk+16+j, 8..15: K=kk+24+j
// B-frag (32x16 bf16): lane L: N=L%16; half j: K = kk + j + 16*(L>=16)
__device__ __forceinline__ v8f qkv_step(const __bf16* __restrict__ xrow,
                                        const __bf16* __restrict__ wrow,
                                        int kk, int half, v8f acc) {
  const __bf16* ap = xrow + kk + (half ? 16 : 0);
  v8bf a0 = *(const v8bf*)(ap);
  v8bf a1 = *(const v8bf*)(ap + 8);
  v16bf af;
#pragma unroll
  for (int j = 0; j < 8; ++j) { af[j] = a0[j]; af[8 + j] = a1[j]; }
  v16bf bf = *(const v16bf*)(wrow + kk + (half ? 16 : 0));
  return __builtin_amdgcn_wmma_f32_16x16x32_bf16(false, af, false, bf,
                                                 (short)0, acc, false, false);
}

// ---------------------------------------------------------------------------
// Kernel 1: qkv = xb @ Wb^T + b, emitted as bf16. Q,K row-major [b][n][d];
// V transposed: Vt[b][d][n] (so attn@V B-fragments load contiguously).
// One wave per 16x16 output tile; 4 waves (4 e-tiles) per block.
// C/D (16x16 f32): VGPR r, lane L: (M = r + 8*(L>=16), N = L%16)
// ---------------------------------------------------------------------------
__global__ __launch_bounds__(128) void qkv_proj_kernel(
    const __bf16* __restrict__ xb, const __bf16* __restrict__ Wb,
    const float* __restrict__ bias,
    __bf16* __restrict__ Qb, __bf16* __restrict__ Kb, __bf16* __restrict__ Vt) {
  const int lane = threadIdx.x & 31;
  const int wave = threadIdx.x >> 5;
  const int l16  = lane & 15;
  const int half = lane >> 4;
  const int et = blockIdx.x * 4 + wave;   // 0..95
  const int n0 = blockIdx.y * 16;
  const int b  = blockIdx.z;
  const int e0 = et * 16;

  const __bf16* xrow = xb + ((size_t)b * SEQ + (size_t)(n0 + l16)) * DMODEL;
  const __bf16* wrow = Wb + (size_t)(e0 + l16) * DMODEL;

  // Two independent accumulator chains (independent back-to-back WMMAs).
  v8f acc0 = {}, acc1 = {};
#pragma unroll 4
  for (int kk = 0; kk < DMODEL; kk += 64) {
    acc0 = qkv_step(xrow, wrow, kk,      half, acc0);
    acc1 = qkv_step(xrow, wrow, kk + 32, half, acc1);
  }
  const float bv = bias[e0 + l16];
  v8f acc;
#pragma unroll
  for (int r = 0; r < 8; ++r) acc[r] = acc0[r] + acc1[r] + bv;

  if (e0 < DMODEL) {
    __bf16* dst = Qb + ((size_t)b * SEQ + (size_t)(n0 + 8 * half)) * DMODEL + (e0 + l16);
#pragma unroll
    for (int r = 0; r < 8; ++r) dst[(size_t)r * DMODEL] = (__bf16)acc[r];
  } else if (e0 < 2 * DMODEL) {
    __bf16* dst = Kb + ((size_t)b * SEQ + (size_t)(n0 + 8 * half)) * DMODEL + (e0 - DMODEL + l16);
#pragma unroll
    for (int r = 0; r < 8; ++r) dst[(size_t)r * DMODEL] = (__bf16)acc[r];
  } else {
    const int dcol = e0 - 2 * DMODEL + l16;
    v8bf tv;
#pragma unroll
    for (int r = 0; r < 8; ++r) tv[r] = (__bf16)acc[r];
    *(v8bf*)(Vt + ((size_t)b * DMODEL + dcol) * SEQ + (n0 + 8 * half)) = tv;  // 16B store
  }
}

// ---------------------------------------------------------------------------
// Kernel 2: flash attention. Block = 4 waves = 16 query rows per block.
// Wave w owns output d-slice [w*128, w*128+128). Per 32-wide k-tile:
//   - each wave: partial QK^T over its 128-d slice (8 WMMAs, s0/s1 interleaved)
//   - partials summed via LDS -> full scores in every wave
//   - online softmax (row reductions via shfl_xor within 16-lane halves)
//   - P (bf16) staged through LDS to re-layout C-frag -> A-frag
//   - 8 independent attn@V WMMAs per wave from contiguous Vt loads
// ---------------------------------------------------------------------------
__global__ __launch_bounds__(128) void flash_attn_kernel(
    const __bf16* __restrict__ Qb, const __bf16* __restrict__ Kb,
    const __bf16* __restrict__ Vt, float* __restrict__ out) {
  const int lane = threadIdx.x & 31;
  const int wave = threadIdx.x >> 5;
  const int l16  = lane & 15;
  const int half = lane >> 4;
  const int q0 = blockIdx.x * 16;
  const int b  = blockIdx.y;
  const int d0 = wave * 128;

  __shared__ float  sc[4][2][32][8];   // per-wave partial score fragments (8KB)
  __shared__ __bf16 pl[16][32];        // softmaxed P tile, row-major (1KB)

  // Preload Q A-fragments for this wave's d-slice (4 x 32-wide K chunks).
  v16bf qa[4];
  {
    const __bf16* qrow = Qb + ((size_t)b * SEQ + (size_t)(q0 + l16)) * DMODEL + d0;
#pragma unroll
    for (int kk = 0; kk < 4; ++kk) {
      const __bf16* p = qrow + kk * 32 + (half ? 16 : 0);
      v8bf x0 = *(const v8bf*)(p);
      v8bf x1 = *(const v8bf*)(p + 8);
#pragma unroll
      for (int j = 0; j < 8; ++j) { qa[kk][j] = x0[j]; qa[kk][8 + j] = x1[j]; }
    }
  }

  v8f zero = {};
  v8f oacc[8];
#pragma unroll
  for (int f = 0; f < 8; ++f) oacc[f] = zero;
  float m_i[8], l_i[8];
#pragma unroll
  for (int r = 0; r < 8; ++r) { m_i[r] = -1.0e30f; l_i[r] = 0.0f; }

  const float scale = 0.125f;  // (D/HEAD_NUMB)^-0.5 = 64^-0.5

  for (int kt = 0; kt < SEQ / 32; ++kt) {
    // ---- partial scores over this wave's 128-d slice (interleaved chains) ----
    v8f s0 = {}, s1 = {};
    {
      const __bf16* k0 = Kb + ((size_t)b * SEQ + (size_t)(kt * 32 + l16)) * DMODEL
                            + d0 + half * 16;
      const __bf16* k1 = k0 + (size_t)16 * DMODEL;  // score cols 16..31
#pragma unroll
      for (int kk = 0; kk < 4; ++kk) {
        v16bf kf0 = *(const v16bf*)(k0 + kk * 32);
        s0 = __builtin_amdgcn_wmma_f32_16x16x32_bf16(false, qa[kk], false, kf0,
                                                     (short)0, s0, false, false);
        v16bf kf1 = *(const v16bf*)(k1 + kk * 32);
        s1 = __builtin_amdgcn_wmma_f32_16x16x32_bf16(false, qa[kk], false, kf1,
                                                     (short)0, s1, false, false);
      }
      // Prefetch next k-tile's K rows (advance = 32 rows * 1KB = 32KB).
      if (kt + 1 < SEQ / 32) {
        __builtin_prefetch(k0 + (size_t)32 * DMODEL, 0, 0);
        __builtin_prefetch(k1 + (size_t)32 * DMODEL, 0, 0);
      }
    }
    *(v8f*)&sc[wave][0][lane][0] = s0;
    *(v8f*)&sc[wave][1][lane][0] = s1;
    __syncthreads();

    // ---- sum partials -> full scores, scaled ----
    float fs0[8], fs1[8];
#pragma unroll
    for (int r = 0; r < 8; ++r) {
      fs0[r] = (sc[0][0][lane][r] + sc[1][0][lane][r] +
                sc[2][0][lane][r] + sc[3][0][lane][r]) * scale;
      fs1[r] = (sc[0][1][lane][r] + sc[1][1][lane][r] +
                sc[2][1][lane][r] + sc[3][1][lane][r]) * scale;
    }

    // ---- online softmax (rows live in 16-lane halves) ----
#pragma unroll
    for (int r = 0; r < 8; ++r) {
      float mx = fmaxf(fs0[r], fs1[r]);
      mx = fmaxf(mx, __shfl_xor(mx, 1, 32));
      mx = fmaxf(mx, __shfl_xor(mx, 2, 32));
      mx = fmaxf(mx, __shfl_xor(mx, 4, 32));
      mx = fmaxf(mx, __shfl_xor(mx, 8, 32));
      float mnew  = fmaxf(m_i[r], mx);
      float alpha = __expf(m_i[r] - mnew);
      m_i[r] = mnew;
      l_i[r] *= alpha;
#pragma unroll
      for (int f = 0; f < 8; ++f) oacc[f][r] *= alpha;
      float p0 = __expf(fs0[r] - mnew);
      float p1 = __expf(fs1[r] - mnew);
      fs0[r] = p0; fs1[r] = p1;
      float rs = p0 + p1;
      rs += __shfl_xor(rs, 1, 32);
      rs += __shfl_xor(rs, 2, 32);
      rs += __shfl_xor(rs, 4, 32);
      rs += __shfl_xor(rs, 8, 32);
      l_i[r] += rs;
    }

    // ---- stage P through LDS (C-layout -> row major) ----
    if (wave == 0) {
#pragma unroll
      for (int r = 0; r < 8; ++r) {
        pl[8 * half + r][l16]      = (__bf16)fs0[r];
        pl[8 * half + r][16 + l16] = (__bf16)fs1[r];
      }
    }
    __syncthreads();

    // ---- read P as A-fragment; attn @ V for this wave's d-slice ----
    v16bf pa;
    {
      const __bf16* prow = &pl[l16][half * 16];
      v8bf x0 = *(const v8bf*)(prow);
      v8bf x1 = *(const v8bf*)(prow + 8);
#pragma unroll
      for (int j = 0; j < 8; ++j) { pa[j] = x0[j]; pa[8 + j] = x1[j]; }
    }
    const __bf16* vrow = Vt + ((size_t)b * DMODEL + (size_t)(d0 + l16)) * SEQ
                            + kt * 32 + half * 16;
#pragma unroll
    for (int f = 0; f < 8; ++f) {   // 8 independent accumulator chains
      v16bf vf = *(const v16bf*)(vrow + (size_t)(16 * f) * SEQ);
      oacc[f] = __builtin_amdgcn_wmma_f32_16x16x32_bf16(false, pa, false, vf,
                                                        (short)0, oacc[f], false, false);
    }
    // cross-iteration LDS hazards are ordered by the two barriers above
  }

  // ---- epilogue: divide by l, store f32 output ----
  float inv[8];
#pragma unroll
  for (int r = 0; r < 8; ++r) inv[r] = 1.0f / l_i[r];
  float* orow = out + ((size_t)b * SEQ + (size_t)(q0 + 8 * half)) * DMODEL + d0 + l16;
#pragma unroll
  for (int f = 0; f < 8; ++f)
#pragma unroll
    for (int r = 0; r < 8; ++r)
      orow[(size_t)r * DMODEL + 16 * f] = oacc[f][r] * inv[r];
}

// ---------------------------------------------------------------------------
extern "C" void kernel_launch(void* const* d_in, const int* in_sizes, int n_in,
                              void* d_out, int out_size, void* d_ws, size_t ws_size,
                              hipStream_t stream) {
  const float* x    = (const float*)d_in[0];
  const float* W    = (const float*)d_in[1];
  const float* bias = (const float*)d_in[2];
  float* out = (float*)d_out;

  // Workspace (bf16): Q, K (row-major), Vt (transposed V), xb, Wb = 65.5 MB.
  __bf16* Qb = (__bf16*)d_ws;
  __bf16* Kb = Qb + (size_t)BATCH * SEQ * DMODEL;
  __bf16* Vt = Kb + (size_t)BATCH * SEQ * DMODEL;
  __bf16* xb = Vt + (size_t)BATCH * SEQ * DMODEL;
  __bf16* Wb = xb + (size_t)BATCH * SEQ * DMODEL;

  // Pre-convert x and W to bf16 (one pass each).
  {
    int n8x = (BATCH * SEQ * DMODEL) / 8;        // 1,048,576
    cvt_bf16_kernel<<<(n8x + 255) / 256, 256, 0, stream>>>(x, xb, n8x);
    int n8w = (EOUT * DMODEL) / 8;               // 98,304
    cvt_bf16_kernel<<<(n8w + 255) / 256, 256, 0, stream>>>(W, Wb, n8w);
  }

  dim3 g1(EOUT / 64, SEQ / 16, BATCH);   // 24 x 256 x 4 blocks, 4 waves each
  qkv_proj_kernel<<<g1, dim3(128, 1, 1), 0, stream>>>(xb, Wb, bias, Qb, Kb, Vt);

  dim3 g2(SEQ / 16, BATCH, 1);           // 256 x 4 blocks, 4 waves each
  flash_attn_kernel<<<g2, dim3(128, 1, 1), 0, stream>>>(Qb, Kb, Vt, out);
}
